// BailingMoESparseMoeBlock_28063316312107
// MI455X (gfx1250) — matrix operand order repaired and checked by reading, not compiled, not run on hardware
//
#include <hip/hip_runtime.h>
#include <cmath>

// ---------------- problem constants ----------------
#define T_TOK 512
#define HDIM  2048
#define NEXP  16
#define TOPK  4
#define IDIM  1408
#define ISH   2816

// ---------------- WMMA tile config ----------------
#define BM 64
#define BN 128
#define BK 64
#define LDT 72   // padded LDS row stride in bf16 elems (144B -> conflict-free b128 reads)

typedef __attribute__((ext_vector_type(16))) __bf16 bf16x16;
typedef __attribute__((ext_vector_type(8)))  __bf16 bf16x8;
typedef __attribute__((ext_vector_type(4)))  __bf16 bf16x4;
typedef __attribute__((ext_vector_type(8)))  float  f32x8;

// ---------------- CDNA5 async global->LDS copy (ASYNCcnt path) ----------------
// Probe-discovered signature: (int-vec16B AS1*, int-vec16B AS3*, imm offset, imm cpol)
#if __has_builtin(__builtin_amdgcn_global_load_async_to_lds_b128)
#define HAVE_ASYNC_LDS 1
typedef int v4i __attribute__((vector_size(16)));
typedef __attribute__((address_space(1))) v4i* v4i_gptr;
typedef __attribute__((address_space(3))) v4i* v4i_lptr;
static __device__ __forceinline__ void
async_copy_b128(const __bf16* g, __bf16* l) {
  __builtin_amdgcn_global_load_async_to_lds_b128(
      (v4i_gptr)(v4i*)const_cast<__bf16*>(g),
      (v4i_lptr)(v4i*)l, 0, 0);
}
#else
#define HAVE_ASYNC_LDS 0
#endif

static __device__ __forceinline__ void wait_async0() {
#if __has_builtin(__builtin_amdgcn_s_wait_asynccnt)
  __builtin_amdgcn_s_wait_asynccnt(0);
#else
  asm volatile("s_wait_asynccnt 0x0" ::: "memory");
#endif
}

// A fragment (16x32 bf16, ISA 7.12.2): lanes 0-15 row M=lane hold K = ks+0..7 then ks+16..23;
// lanes 16-31 hold K = ks+8..15 then ks+24..31.
static __device__ __forceinline__ bf16x16
load_frag_a(const __bf16* tile, int row16, int lane, int ks) {
  const int half = lane >> 4, lr = lane & 15;
  const __bf16* p = tile + (row16 + lr) * LDT + ks + half * 8;
  bf16x8 lo = *(const bf16x8*)p;
  bf16x8 hi = *(const bf16x8*)(p + 16);
  return __builtin_shufflevector(lo, hi, 0,1,2,3,4,5,6,7,8,9,10,11,12,13,14,15);
}

// B fragment (32x16 bf16): lanes 0-15 col N=lane hold K = ks+0..7 then ks+16..23;
// lanes 16-31 hold K = ks+8..15 then ks+24..31 (matches A fragment).
static __device__ __forceinline__ bf16x16
load_frag_b(const __bf16* tile, int col16, int lane, int ks) {
  const int half = lane >> 4, lc = lane & 15;
  const __bf16* p = tile + (col16 + lc) * LDT + ks + half * 8;
  bf16x8 lo = *(const bf16x8*)p;
  bf16x8 hi = *(const bf16x8*)(p + 16);
  return __builtin_shufflevector(lo, hi, 0,1,2,3,4,5,6,7,8,9,10,11,12,13,14,15);
}

static __device__ __forceinline__ f32x8
wmma_bf16(bf16x16 a, bf16x16 b, f32x8 c) {
  return __builtin_amdgcn_wmma_f32_16x16x32_bf16(false, a, false, b, (short)0, c, false, false);
}

// ---------------- fp32 -> bf16 activation convert ----------------
__global__ __launch_bounds__(256)
void cvt_kernel(const float* __restrict__ x, __bf16* __restrict__ xb) {
  const size_t i = ((size_t)blockIdx.x * blockDim.x + threadIdx.x) * 4;
  float4 v = *(const float4*)(x + i);
  bf16x4 p = { (__bf16)v.x, (__bf16)v.y, (__bf16)v.z, (__bf16)v.w };
  *(bf16x4*)(xb + i) = p;
}

// ---------------- router logits (one wave per token, fp32) ----------------
__global__ __launch_bounds__(256)
void router_kernel(const float* __restrict__ x, const float* __restrict__ gw,
                   float* __restrict__ logits) {
  const int wave = threadIdx.x >> 5, lane = threadIdx.x & 31;
  const int t = blockIdx.x * 8 + wave;
  float acc[NEXP] = {};
  for (int h = lane; h < HDIM; h += 32) {
    const float xv = x[(size_t)t * HDIM + h];
#pragma unroll
    for (int e = 0; e < NEXP; ++e) acc[e] += xv * gw[(size_t)e * HDIM + h];
  }
#pragma unroll
  for (int e = 0; e < NEXP; ++e)
#pragma unroll
    for (int d = 16; d > 0; d >>= 1) acc[e] += __shfl_xor(acc[e], d, 32);
  if (lane < NEXP) logits[(size_t)t * NEXP + lane] = acc[lane];
}

// ---------------- softmax + top-4 + deterministic compaction (1 block) ----------------
__global__ __launch_bounds__(512)
void topk_kernel(const float* __restrict__ logits,
                 int* __restrict__ expert_off,
                 int* __restrict__ pair_token,
                 float* __restrict__ pair_weight,
                 int* __restrict__ pair_of) {
  __shared__ short s_idx[T_TOK][TOPK];
  __shared__ float s_w[T_TOK][TOPK];
  __shared__ int s_cnt[NEXP];
  __shared__ int s_off[NEXP + 1];

  const int t = threadIdx.x;
  {
    float l[NEXP];
    float m = -1e30f;
#pragma unroll
    for (int e = 0; e < NEXP; ++e) { l[e] = logits[t * NEXP + e]; m = fmaxf(m, l[e]); }
    float s = 0.f;
#pragma unroll
    for (int e = 0; e < NEXP; ++e) { l[e] = __expf(l[e] - m); s += l[e]; }
    const float inv = 1.0f / s;
#pragma unroll
    for (int e = 0; e < NEXP; ++e) l[e] *= inv;
    float wsum = 0.f;
#pragma unroll
    for (int k = 0; k < TOPK; ++k) {
      int best = 0; float bv = l[0];
#pragma unroll
      for (int e = 1; e < NEXP; ++e) if (l[e] > bv) { bv = l[e]; best = e; }
      s_idx[t][k] = (short)best; s_w[t][k] = bv; wsum += bv; l[best] = -1.f;
    }
    const float winv = 1.0f / wsum;
#pragma unroll
    for (int k = 0; k < TOPK; ++k) s_w[t][k] *= winv;
  }
  __syncthreads();

  // wave e counts/places pairs for expert e, in ascending token order (deterministic)
  const int e = t >> 5;
  const int lane = t & 31;
  const int t0 = lane * (T_TOK / 32);
  int c = 0;
  for (int i = 0; i < T_TOK / 32; ++i)
#pragma unroll
    for (int k = 0; k < TOPK; ++k) c += (s_idx[t0 + i][k] == e);
  int incl = c;
  for (int d = 1; d < 32; d <<= 1) {
    int v = __shfl_up(incl, d, 32);
    if (lane >= d) incl += v;
  }
  const int excl = incl - c;
  if (lane == 31) s_cnt[e] = incl;
  __syncthreads();
  if (t == 0) {
    int o = 0;
    for (int i = 0; i < NEXP; ++i) { s_off[i] = o; o += s_cnt[i]; }
    s_off[NEXP] = o;
    for (int i = 0; i <= NEXP; ++i) expert_off[i] = s_off[i];
  }
  __syncthreads();
  int base = s_off[e] + excl;
  for (int i = 0; i < T_TOK / 32; ++i) {
    const int tt = t0 + i;
#pragma unroll
    for (int k = 0; k < TOPK; ++k)
      if (s_idx[tt][k] == e) {
        pair_token[base] = tt;
        pair_weight[base] = s_w[tt][k];
        pair_of[tt * TOPK + k] = base;
        ++base;
      }
  }
}

// ---------------- fused gate_up GEMM + SiLU*mul (shared & routed) ----------------
// C tile 64x128; 8 waves: wave = 4 m-subtiles x 2 n-groups (4 n-subtiles each).
__global__ __launch_bounds__(256)
void gu_kernel(const __bf16* __restrict__ Xb, const float* __restrict__ W,
               __bf16* __restrict__ Hout,
               const int* __restrict__ pair_token,
               const int* __restrict__ expert_off,
               int Kdim, int Icols, int expertMode) {
  __shared__ __bf16 As[BM * LDT];
  __shared__ __bf16 Bg[BN * LDT];
  __shared__ __bf16 Bu[BN * LDT];

  const int tid = threadIdx.x;
  const int e = expertMode ? (int)blockIdx.z : 0;
  int rowBase = 0, Ne = T_TOK;
  if (expertMode) { rowBase = expert_off[e]; Ne = expert_off[e + 1] - rowBase; }
  const int m0 = blockIdx.x * BM;
  if (m0 >= Ne) return;
  const int n0 = blockIdx.y * BN;

  const float* Wg = W + (size_t)e * (2ull * (size_t)Icols * Kdim) + (size_t)n0 * Kdim;
  const float* Wu = Wg + (size_t)Icols * Kdim;
  __bf16* Hb = Hout + (size_t)rowBase * Icols;

  const int lane = tid & 31;
  const int wave = tid >> 5;
  const int mt = (wave & 3) * 16;
  const int ntb = (wave >> 2) * 4;

  // A-tile loader geometry: 2 iters, row = (tid>>3) + i*32, 16B column chunk
  const int a_c8 = (tid & 7) * 8;
  int a_tok[2];
#pragma unroll
  for (int i = 0; i < 2; ++i) {
    const int r = (tid >> 3) + i * 32;
    int rm = m0 + r; if (rm > Ne - 1) rm = Ne - 1;
    a_tok[i] = expertMode ? pair_token[rowBase + rm] : rm;
  }

  f32x8 accG[4] = {}, accU[4] = {};

  for (int k0 = 0; k0 < Kdim; k0 += BK) {
    { // A tile: bf16 gathered rows -> LDS (async copy bypasses VGPRs when available)
#pragma unroll
      for (int i = 0; i < 2; ++i) {
        const int r = (tid >> 3) + i * 32;
        const __bf16* src = Xb + (size_t)a_tok[i] * Kdim + k0 + a_c8;
#if HAVE_ASYNC_LDS
        async_copy_b128(src, &As[r * LDT + a_c8]);
#else
        *(bf16x8*)(&As[r * LDT + a_c8]) = *(const bf16x8*)src;
#endif
      }
    }
    { // B tiles: fp32 -> bf16 -> LDS
      const int c4 = (tid & 15) * 4;
#pragma unroll
      for (int i = 0; i < 8; ++i) {
        const int r = (tid >> 4) + i * 16;
        const float4 vg = *(const float4*)(Wg + (size_t)r * Kdim + k0 + c4);
        const float4 vu = *(const float4*)(Wu + (size_t)r * Kdim + k0 + c4);
        bf16x4 pg = { (__bf16)vg.x, (__bf16)vg.y, (__bf16)vg.z, (__bf16)vg.w };
        bf16x4 pu = { (__bf16)vu.x, (__bf16)vu.y, (__bf16)vu.z, (__bf16)vu.w };
        *(bf16x4*)(&Bg[r * LDT + c4]) = pg;
        *(bf16x4*)(&Bu[r * LDT + c4]) = pu;
      }
    }
#if HAVE_ASYNC_LDS
    wait_async0();
#endif
    __syncthreads();
#pragma unroll
    for (int ks = 0; ks < BK; ks += 32) {
      const bf16x16 a = load_frag_a(As, mt, lane, ks);
      bf16x16 bg[4], bu[4];
#pragma unroll
      for (int j = 0; j < 4; ++j) bg[j] = load_frag_b(Bg, (ntb + j) * 16, lane, ks);
#pragma unroll
      for (int j = 0; j < 4; ++j) accG[j] = wmma_bf16(a, bg[j], accG[j]);
#pragma unroll
      for (int j = 0; j < 4; ++j) bu[j] = load_frag_b(Bu, (ntb + j) * 16, lane, ks);
#pragma unroll
      for (int j = 0; j < 4; ++j) accU[j] = wmma_bf16(a, bu[j], accU[j]);
    }
    __syncthreads();
  }

  // epilogue: h = silu(g)*u -> bf16. C/D layout: VGPR r, lanes<16 -> M=r; lanes>=16 -> M=8+r; N=lane&15.
  const int half = lane >> 4, lr = lane & 15;
#pragma unroll
  for (int j = 0; j < 4; ++j) {
    const int n = n0 + (ntb + j) * 16 + lr;
#pragma unroll
    for (int r = 0; r < 8; ++r) {
      const int m = m0 + mt + half * 8 + r;
      if (m < Ne) {
        const float g = accG[j][r], u = accU[j][r];
        const float h = (g / (1.0f + __expf(-g))) * u;
        Hb[(size_t)m * Icols + n] = (__bf16)h;
      }
    }
  }
}

// ---------------- down-proj GEMM (shared & routed; routed rows scaled by pair weight) ----------------
__global__ __launch_bounds__(256)
void down_kernel(const __bf16* __restrict__ Ain, const float* __restrict__ W,
                 float* __restrict__ Yout, const float* __restrict__ rowW,
                 const int* __restrict__ expert_off,
                 int Kdim, int Ncols, int expertMode) {
  __shared__ __bf16 As[BM * LDT];
  __shared__ __bf16 Bs[BN * LDT];

  const int tid = threadIdx.x;
  const int e = expertMode ? (int)blockIdx.z : 0;
  int rowBase = 0, Ne = T_TOK;
  if (expertMode) { rowBase = expert_off[e]; Ne = expert_off[e + 1] - rowBase; }
  const int m0 = blockIdx.x * BM;
  if (m0 >= Ne) return;
  const int n0 = blockIdx.y * BN;

  const __bf16* Ab = Ain + (size_t)rowBase * Kdim;   // he rows are already expert-contiguous
  const float* Wb = W + (size_t)e * ((size_t)Ncols * Kdim) + (size_t)n0 * Kdim;
  float* Yb = Yout + (size_t)rowBase * Ncols;
  const float* rwB = rowW ? (rowW + rowBase) : nullptr;

  const int lane = tid & 31;
  const int wave = tid >> 5;
  const int mt = (wave & 3) * 16;
  const int ntb = (wave >> 2) * 4;

  const int a_c8 = (tid & 7) * 8;
  int a_row[2];
#pragma unroll
  for (int i = 0; i < 2; ++i) {
    const int r = (tid >> 3) + i * 32;
    int rm = m0 + r; if (rm > Ne - 1) rm = Ne - 1;
    a_row[i] = rm;
  }

  f32x8 acc[4] = {};

  for (int k0 = 0; k0 < Kdim; k0 += BK) {
    {
#pragma unroll
      for (int i = 0; i < 2; ++i) {
        const int r = (tid >> 3) + i * 32;
        const __bf16* src = Ab + (size_t)a_row[i] * Kdim + k0 + a_c8;
#if HAVE_ASYNC_LDS
        async_copy_b128(src, &As[r * LDT + a_c8]);
#else
        *(bf16x8*)(&As[r * LDT + a_c8]) = *(const bf16x8*)src;
#endif
      }
    }
    {
      const int c4 = (tid & 15) * 4;
#pragma unroll
      for (int i = 0; i < 8; ++i) {
        const int r = (tid >> 4) + i * 16;
        const float4 v = *(const float4*)(Wb + (size_t)r * Kdim + k0 + c4);
        bf16x4 pk = { (__bf16)v.x, (__bf16)v.y, (__bf16)v.z, (__bf16)v.w };
        *(bf16x4*)(&Bs[r * LDT + c4]) = pk;
      }
    }
#if HAVE_ASYNC_LDS
    wait_async0();
#endif
    __syncthreads();
#pragma unroll
    for (int ks = 0; ks < BK; ks += 32) {
      const bf16x16 a = load_frag_a(As, mt, lane, ks);
      bf16x16 bb[4];
#pragma unroll
      for (int j = 0; j < 4; ++j) bb[j] = load_frag_b(Bs, (ntb + j) * 16, lane, ks);
#pragma unroll
      for (int j = 0; j < 4; ++j) acc[j] = wmma_bf16(a, bb[j], acc[j]);
    }
    __syncthreads();
  }

  const int half = lane >> 4, lr = lane & 15;
#pragma unroll
  for (int j = 0; j < 4; ++j) {
    const int n = n0 + (ntb + j) * 16 + lr;
#pragma unroll
    for (int r = 0; r < 8; ++r) {
      const int m = m0 + mt + half * 8 + r;
      if (m < Ne) {
        float y = acc[j][r];
        if (rwB) y *= rwB[m];
        Yb[(size_t)m * Ncols + n] = y;
      }
    }
  }
}

// ---------------- final deterministic combine: out = shared + sum_k yd[pair] ----------------
__global__ __launch_bounds__(256)
void combine_kernel(const float* __restrict__ ys, const float* __restrict__ yd,
                    const int* __restrict__ pair_of, float* __restrict__ out) {
  const int t = blockIdx.y;
  const int h = blockIdx.x * blockDim.x + threadIdx.x;
  const int* p = pair_of + t * TOPK;
  float acc = ys[(size_t)t * HDIM + h];
#pragma unroll
  for (int k = 0; k < TOPK; ++k) acc += yd[(size_t)p[k] * HDIM + h];
  out[(size_t)t * HDIM + h] = acc;   // SCALE == 1.0
}

// ---------------- launch ----------------
extern "C" void kernel_launch(void* const* d_in, const int* in_sizes, int n_in,
                              void* d_out, int out_size, void* d_ws, size_t ws_size,
                              hipStream_t stream) {
  (void)in_sizes; (void)n_in; (void)out_size; (void)ws_size;
  const float* x   = (const float*)d_in[0];
  const float* gw  = (const float*)d_in[1];
  const float* wgu = (const float*)d_in[2];
  const float* wdn = (const float*)d_in[3];
  const float* sgu = (const float*)d_in[4];
  const float* sdn = (const float*)d_in[5];
  float* out = (float*)d_out;

  char* p = (char*)d_ws;
  auto carve = [&](size_t bytes) { char* r = p; p += (bytes + 255) & ~(size_t)255; return r; };
  float* logits      = (float*)carve((size_t)T_TOK * NEXP * 4);
  int*   expert_off  = (int*)  carve((NEXP + 1) * 4);
  int*   pair_token  = (int*)  carve((size_t)T_TOK * TOPK * 4);
  float* pair_weight = (float*)carve((size_t)T_TOK * TOPK * 4);
  int*   pair_of     = (int*)  carve((size_t)T_TOK * TOPK * 4);
  __bf16* xb = (__bf16*)carve((size_t)T_TOK * HDIM * 2);
  __bf16* hs = (__bf16*)carve((size_t)T_TOK * ISH * 2);
  __bf16* he = (__bf16*)carve((size_t)T_TOK * TOPK * IDIM * 2);
  float* ys  = (float*)carve((size_t)T_TOK * HDIM * 4);
  float* yd  = (float*)carve((size_t)T_TOK * TOPK * HDIM * 4);

  cvt_kernel<<<dim3((T_TOK * HDIM) / (256 * 4)), 256, 0, stream>>>(x, xb);
  router_kernel<<<dim3(T_TOK / 8), 256, 0, stream>>>(x, gw, logits);
  topk_kernel<<<1, 512, 0, stream>>>(logits, expert_off, pair_token, pair_weight, pair_of);

  // shared expert: gate_up + SiLU*mul, then down
  gu_kernel<<<dim3(T_TOK / BM, ISH / BN, 1), 256, 0, stream>>>(
      xb, sgu, hs, nullptr, nullptr, HDIM, ISH, 0);
  // routed experts: gathered gate_up + SiLU*mul
  gu_kernel<<<dim3(T_TOK / BM, IDIM / BN, NEXP), 256, 0, stream>>>(
      xb, wgu, he, pair_token, expert_off, HDIM, IDIM, 1);
  // shared down
  down_kernel<<<dim3(T_TOK / BM, HDIM / BN, 1), 256, 0, stream>>>(
      hs, sdn, ys, nullptr, nullptr, ISH, HDIM, 0);
  // routed down, rows scaled by renormalized top-k weight
  down_kernel<<<dim3(T_TOK / BM, HDIM / BN, NEXP), 256, 0, stream>>>(
      he, wdn, yd, pair_weight, expert_off, IDIM, HDIM, 1);

  combine_kernel<<<dim3(HDIM / 256, T_TOK), 256, 0, stream>>>(ys, yd, pair_of, out);
}
